// Attention_512_64699387347343
// MI455X (gfx1250) — matrix operand order repaired and checked by reading, not compiled, not run on hardware
//
#include <hip/hip_runtime.h>
#include <cstdint>
#include <cstddef>

// ---------------------------------------------------------------------------
// Problem constants (fixed by the reference)
// ---------------------------------------------------------------------------
#define GG 128
#define SS 2048
#define LL 256
#define FF 512
#define DD 512

typedef __attribute__((ext_vector_type(16))) __bf16 v16bf;
typedef __attribute__((ext_vector_type(8)))  float  v8f;

__device__ __forceinline__ unsigned short f2bf(float f) {
    unsigned int u = __builtin_bit_cast(unsigned int, f);
    u += 0x7FFFu + ((u >> 16) & 1u);           // round-to-nearest-even
    return (unsigned short)(u >> 16);
}
__device__ __forceinline__ float bf2f(unsigned short s) {
    unsigned int u = ((unsigned int)s) << 16;
    return __builtin_bit_cast(float, u);
}

// ---------------------------------------------------------------------------
// Elementwise f32 -> bf16
// ---------------------------------------------------------------------------
__global__ void __launch_bounds__(256) conv_bf16_kernel(
    const float* __restrict__ in, unsigned short* __restrict__ out, int n) {
    int i = blockIdx.x * 256 + threadIdx.x;
    if (i < n) out[i] = f2bf(in[i]);
}

// Transposed convert: W[K,N] f32 (row-major)  ->  WT[N,K] bf16
__global__ void __launch_bounds__(256) tconv_bf16_kernel(
    const float* __restrict__ W, unsigned short* __restrict__ WT, int K, int N) {
    int i = blockIdx.x * 256 + threadIdx.x;
    if (i >= K * N) return;
    int k = i / N, n = i - k * N;
    WT[(size_t)n * K + k] = f2bf(W[i]);
}

// V[G,L,D] bf16 -> Vt[G,D,L] bf16
__global__ void __launch_bounds__(256) transpose_v_kernel(
    const unsigned short* __restrict__ V, unsigned short* __restrict__ Vt) {
    size_t idx = (size_t)blockIdx.x * 256 + threadIdx.x;   // G*L*D elements
    int d = (int)(idx & (DD - 1));
    int m = (int)((idx >> 9) & (LL - 1));
    int g = (int)(idx >> 17);
    Vt[((size_t)g << 17) + ((size_t)d << 8) + m] = V[idx];
}

// ---------------------------------------------------------------------------
// Fragment bundle: one A 16x32 fragment + four B 32x16 fragments.
// Layouts per CDNA5 ISA 7.12.2 (16-bit A 16x32 / B 32x16, wave32).
// ---------------------------------------------------------------------------
struct FragSet {
    union { v16bf v; unsigned int u[8]; } a;
    union { v16bf v; unsigned int u[8]; } b[4];
};

__device__ __forceinline__ void load_frags(
    FragSet& f, const unsigned short* __restrict__ Atile,
    const unsigned short* __restrict__ Btile, size_t bsub, int k0, int h) {
#pragma unroll
    for (int j = 0; j < 8; ++j) {
        // A 16x32 bf16 fragment: VGPR j holds K = base + {0,1}
        int ka = ((j < 4) ? 0 : 16) + 8 * h + 2 * (j & 3);
        f.a.u[j] = *reinterpret_cast<const unsigned int*>(Atile + k0 + ka);
        // B 32x16 bf16 fragment (Bt row-major NxK): VGPR j holds same K as A
        int kb = ((j < 4) ? 0 : 16) + 8 * h + 2 * (j & 3);
#pragma unroll
        for (int q = 0; q < 4; ++q)
            f.b[q].u[j] = *reinterpret_cast<const unsigned int*>(
                Btile + q * bsub + k0 + kb);
    }
}

// ---------------------------------------------------------------------------
// Generic batched bf16 WMMA GEMM, 16x64 tile per wave:
//   4 independent accumulators -> 4 back-to-back v_wmma_f32_16x16x32_bf16
//   per 32-k step, A-fragment reused 4x, 2-deep software pipeline so the
//   next k-step's VMEM loads are in flight while the current WMMAs execute.
//   C[g] (MxN) = A[g] (MxK, row-major bf16) x Bt[g]^T   (Bt is NxK row-major)
//   optional f32 bias per column; output bf16 or f32.
// Requires: M % 16 == 0, N % 64 == 0, K % 64 == 0, K >= 64 (true here).
// ---------------------------------------------------------------------------
__global__ void __launch_bounds__(256) gemm_bf16_wmma(
    const unsigned short* __restrict__ A,  int lda, long long sA,
    const unsigned short* __restrict__ Bt, int ldb, long long sB,
    void* __restrict__ Cout,               int ldc, long long sC,
    int M, int N, int K,
    const float* __restrict__ bias, int outBf16) {
    int g = blockIdx.y;
    A  += (size_t)g * sA;
    Bt += (size_t)g * sB;

    int lane = threadIdx.x & 31;
    int wv   = threadIdx.x >> 5;
    int tilesM  = M >> 4;
    int tilesNG = N >> 6;                        // 64-wide n-groups
    int t = blockIdx.x * 8 + wv;
    if (t >= tilesM * tilesNG) return;           // wave-uniform: EXEC stays full
    int tm = t % tilesM;                         // tm fastest: block shares B strip
    int tg = t / tilesM;

    int h  = lane >> 4;          // half-wave select
    int ln = lane & 15;          // row of A / col of B (both use lane%16)

    const unsigned short* Atile = A  + (size_t)(tm * 16 + ln) * lda;
    const unsigned short* Btile = Bt + (size_t)(tg * 64 + ln) * ldb;
    const size_t bsub = (size_t)16 * ldb;        // next 16-column sub-tile of B

    v8f acc[4] = {{}, {}, {}, {}};
    FragSet f0, f1;

    // 2-deep pipeline: loads for step n+1 issue before WMMAs of step n retire.
    load_frags(f0, Atile, Btile, bsub, 0, h);
    int k0 = 0;
    for (; k0 < K - 64; k0 += 64) {
        load_frags(f1, Atile, Btile, bsub, k0 + 32, h);
#pragma unroll
        for (int q = 0; q < 4; ++q)
            acc[q] = __builtin_amdgcn_wmma_f32_16x16x32_bf16(
                false, f0.a.v, false, f0.b[q].v, (short)0, acc[q], false, false);
        load_frags(f0, Atile, Btile, bsub, k0 + 64, h);
#pragma unroll
        for (int q = 0; q < 4; ++q)
            acc[q] = __builtin_amdgcn_wmma_f32_16x16x32_bf16(
                false, f1.a.v, false, f1.b[q].v, (short)0, acc[q], false, false);
    }
    // tail: k0 == K-64, two remaining 32-k steps
    load_frags(f1, Atile, Btile, bsub, k0 + 32, h);
#pragma unroll
    for (int q = 0; q < 4; ++q)
        acc[q] = __builtin_amdgcn_wmma_f32_16x16x32_bf16(
            false, f0.a.v, false, f0.b[q].v, (short)0, acc[q], false, false);
#pragma unroll
    for (int q = 0; q < 4; ++q)
        acc[q] = __builtin_amdgcn_wmma_f32_16x16x32_bf16(
            false, f1.a.v, false, f1.b[q].v, (short)0, acc[q], false, false);

#pragma unroll
    for (int q = 0; q < 4; ++q) {
        int ncol = tg * 64 + q * 16 + ln;
        float bval = bias ? bias[ncol] : 0.0f;
        if (outBf16) {
            unsigned short* C = (unsigned short*)Cout + (size_t)g * sC;
#pragma unroll
            for (int r = 0; r < 8; ++r) {
                int m = tm * 16 + r + 8 * h;     // C layout: VGPR r -> M = r + 8h
                C[(size_t)m * ldc + ncol] = f2bf(acc[q][r] + bval);
            }
        } else {
            float* C = (float*)Cout + (size_t)g * sC;
#pragma unroll
            for (int r = 0; r < 8; ++r) {
                int m = tm * 16 + r + 8 * h;
                C[(size_t)m * ldc + ncol] = acc[q][r] + bval;
            }
        }
    }
}

// ---------------------------------------------------------------------------
// Masked row softmax over 256 keys; one wave32 per row. f32 in, bf16 out.
// ---------------------------------------------------------------------------
__global__ void __launch_bounds__(256) softmax_mask_kernel(
    const float* __restrict__ scores, unsigned short* __restrict__ attn,
    const int* __restrict__ lengths) {
    int lane = threadIdx.x & 31;
    int wv   = threadIdx.x >> 5;
    int row  = blockIdx.x * 8 + wv;              // [0, G*L)
    int len  = lengths[row >> 8];
    const float* srow = scores + (size_t)row * LL;

    float v[8];
    float mx = -__builtin_inff();
#pragma unroll
    for (int i = 0; i < 8; ++i) {
        int m = lane + 32 * i;
        float s = srow[m];
        v[i] = (m < len) ? s : -__builtin_inff();
        mx = fmaxf(mx, v[i]);
    }
    for (int off = 16; off >= 1; off >>= 1) mx = fmaxf(mx, __shfl_xor(mx, off, 32));

    float e[8];
    float sum = 0.0f;
#pragma unroll
    for (int i = 0; i < 8; ++i) { e[i] = __expf(v[i] - mx); sum += e[i]; }
    for (int off = 16; off >= 1; off >>= 1) sum += __shfl_xor(sum, off, 32);

    float inv = 1.0f / sum;
#pragma unroll
    for (int i = 0; i < 8; ++i)
        attn[(size_t)row * LL + lane + 32 * i] = f2bf(e[i] * inv);
}

// ---------------------------------------------------------------------------
// w[g,l] = h3[g,l,:]·W4 + b4, then length==1 onehot case + key mask.
// One wave32 per row (64-length dot).
// ---------------------------------------------------------------------------
__global__ void __launch_bounds__(256) score_w_kernel(
    const unsigned short* __restrict__ h3, const float* __restrict__ W4,
    const float* __restrict__ b4, const int* __restrict__ lengths,
    float* __restrict__ w) {
    int lane = threadIdx.x & 31;
    int wv   = threadIdx.x >> 5;
    int row  = blockIdx.x * 8 + wv;              // [0, G*L)
    const unsigned short* hr = h3 + (size_t)row * 64;
    float acc = bf2f(hr[lane]) * W4[lane] + bf2f(hr[lane + 32]) * W4[lane + 32];
    for (int off = 16; off >= 1; off >>= 1) acc += __shfl_xor(acc, off, 32);
    if (lane == 0) {
        int g = row >> 8, l = row & (LL - 1);
        int len = lengths[g];
        float val = acc + b4[0];
        if (len == 1)      val = (l == 0) ? 1.0f : 0.0f;
        else if (l >= len) val = 0.0f;
        w[row] = val;
    }
}

// ---------------------------------------------------------------------------
// out[s,g] = sum_l raw[g,s,l] * w[g,l]  — memory-bound, coalesced raw reads.
// grid = (S/8, G); one wave32 per sample row; w[g,:] staged in LDS.
// ---------------------------------------------------------------------------
__global__ void __launch_bounds__(256) weighted_sum_kernel(
    const float* __restrict__ raw, const float* __restrict__ w,
    float* __restrict__ out) {
    __shared__ float wl[LL];
    int g = blockIdx.y;
    wl[threadIdx.x] = w[(g << 8) + threadIdx.x];
    __syncthreads();
    int lane = threadIdx.x & 31;
    int wv   = threadIdx.x >> 5;
    int s    = blockIdx.x * 8 + wv;
    const float* rrow = raw + ((size_t)g * SS + s) * LL;
    __builtin_prefetch(rrow + 8 * LL, 0, 0);     // next s-block (global_prefetch_b8)
    float acc = 0.0f;
#pragma unroll
    for (int i = 0; i < 8; ++i) {
        int l = lane + 32 * i;
        acc += rrow[l] * wl[l];
    }
    for (int off = 16; off >= 1; off >>= 1) acc += __shfl_xor(acc, off, 32);
    if (lane == 0) out[(size_t)s * GG + g] = acc;
}

// ---------------------------------------------------------------------------
// Host-side orchestration
// ---------------------------------------------------------------------------
extern "C" void kernel_launch(void* const* d_in, const int* in_sizes, int n_in,
                              void* d_out, int out_size, void* d_ws, size_t ws_size,
                              hipStream_t stream) {
    const float* raw  = (const float*)d_in[0];
    const float* info = (const float*)d_in[1];
    const float* Wq   = (const float*)d_in[2];
    const float* Wk   = (const float*)d_in[3];
    const float* Wv   = (const float*)d_in[4];
    const float* W1   = (const float*)d_in[5];
    const float* b1   = (const float*)d_in[6];
    const float* W2   = (const float*)d_in[7];
    const float* b2   = (const float*)d_in[8];
    const float* W3   = (const float*)d_in[9];
    const float* b3   = (const float*)d_in[10];
    const float* W4   = (const float*)d_in[11];
    const float* b4   = (const float*)d_in[12];
    const int*   len  = (const int*)d_in[13];
    float* out = (float*)d_out;

    // --- workspace layout (bytes) ---
    constexpr size_t OFF_INFO  = 0;                        // 33.5 MB bf16 info
    constexpr size_t OFF_WQT   = 33554432;                 // 512x512 bf16
    constexpr size_t OFF_WKT   = OFF_WQT + 524288;
    constexpr size_t OFF_WVT   = OFF_WKT + 524288;
    constexpr size_t OFF_W1T   = OFF_WVT + 524288;         // [256,512]
    constexpr size_t OFF_W2T   = OFF_W1T + 262144;         // [128,256]
    constexpr size_t OFF_W3T   = OFF_W2T + 65536;          // [64,128]
    constexpr size_t OFF_Q     = 35651584;                 // 33.5 MB
    constexpr size_t OFF_K     = OFF_Q  + 33554432;
    constexpr size_t OFF_V     = OFF_K  + 33554432;
    constexpr size_t OFF_VT    = OFF_V  + 33554432;
    constexpr size_t OFF_SC    = OFF_VT + 33554432;        // scores f32, 33.5 MB
    // aliases (lifetimes disjoint):
    constexpr size_t OFF_CTX   = OFF_Q;                    // ctx  over Q
    constexpr size_t OFF_ATTN  = OFF_V;                    // attn over V
    constexpr size_t OFF_H1    = OFF_K;                    // h1   over K
    constexpr size_t OFF_H2    = OFF_K + 16777216;
    constexpr size_t OFF_H3    = OFF_SC;                   // h3   over scores
    constexpr size_t OFF_W     = OFF_SC + 4194304;         // w vec f32

    char* ws = (char*)d_ws;
    unsigned short* infoB = (unsigned short*)(ws + OFF_INFO);
    unsigned short* WqT = (unsigned short*)(ws + OFF_WQT);
    unsigned short* WkT = (unsigned short*)(ws + OFF_WKT);
    unsigned short* WvT = (unsigned short*)(ws + OFF_WVT);
    unsigned short* W1T = (unsigned short*)(ws + OFF_W1T);
    unsigned short* W2T = (unsigned short*)(ws + OFF_W2T);
    unsigned short* W3T = (unsigned short*)(ws + OFF_W3T);
    unsigned short* Qb  = (unsigned short*)(ws + OFF_Q);
    unsigned short* Kb  = (unsigned short*)(ws + OFF_K);
    unsigned short* Vb  = (unsigned short*)(ws + OFF_V);
    unsigned short* Vt  = (unsigned short*)(ws + OFF_VT);
    float*          sc  = (float*)(ws + OFF_SC);
    unsigned short* at  = (unsigned short*)(ws + OFF_ATTN);
    unsigned short* ctx = (unsigned short*)(ws + OFF_CTX);
    unsigned short* h1  = (unsigned short*)(ws + OFF_H1);
    unsigned short* h2  = (unsigned short*)(ws + OFF_H2);
    unsigned short* h3  = (unsigned short*)(ws + OFF_H3);
    float*          wv  = (float*)(ws + OFF_W);

    const int M = GG * LL;              // 32768
    // tiles = (M/16) * (N/64); 8 waves (tiles) per block
    auto blocksFor = [](int M_, int N_) { return ((M_ >> 4) * (N_ >> 6) + 7) / 8; };

    // 1. convert info + weights to bf16 (weights transposed)
    conv_bf16_kernel<<<(M * FF) / 256, 256, 0, stream>>>(info, infoB, M * FF);
    tconv_bf16_kernel<<<(FF * DD + 255) / 256, 256, 0, stream>>>(Wq, WqT, FF, DD);
    tconv_bf16_kernel<<<(FF * DD + 255) / 256, 256, 0, stream>>>(Wk, WkT, FF, DD);
    tconv_bf16_kernel<<<(FF * DD + 255) / 256, 256, 0, stream>>>(Wv, WvT, FF, DD);
    tconv_bf16_kernel<<<(DD * 256 + 255) / 256, 256, 0, stream>>>(W1, W1T, DD, 256);
    tconv_bf16_kernel<<<(256 * 128 + 255) / 256, 256, 0, stream>>>(W2, W2T, 256, 128);
    tconv_bf16_kernel<<<(128 * 64 + 255) / 256, 256, 0, stream>>>(W3, W3T, 128, 64);

    // 2. QKV projections: [32768,512] x [512,512]
    gemm_bf16_wmma<<<dim3(blocksFor(M, DD), 1), 256, 0, stream>>>(
        infoB, FF, 0, WqT, FF, 0, Qb, DD, 0, M, DD, FF, nullptr, 1);
    gemm_bf16_wmma<<<dim3(blocksFor(M, DD), 1), 256, 0, stream>>>(
        infoB, FF, 0, WkT, FF, 0, Kb, DD, 0, M, DD, FF, nullptr, 1);
    gemm_bf16_wmma<<<dim3(blocksFor(M, DD), 1), 256, 0, stream>>>(
        infoB, FF, 0, WvT, FF, 0, Vb, DD, 0, M, DD, FF, nullptr, 1);

    // 3. V -> Vt  [G,L,D] -> [G,D,L]
    transpose_v_kernel<<<(M * DD) / 256, 256, 0, stream>>>(Vb, Vt);

    // 4. scores[g] = Q_g K_g^T  (Bt == K_g row-major [m,d]); f32 out
    gemm_bf16_wmma<<<dim3(blocksFor(LL, LL), GG), 256, 0, stream>>>(
        Qb, DD, (long long)LL * DD, Kb, DD, (long long)LL * DD,
        sc, LL, (long long)LL * LL, LL, LL, DD, nullptr, 0);

    // 5. masked softmax -> bf16 attn
    softmax_mask_kernel<<<M / 8, 256, 0, stream>>>(sc, at, len);

    // 6. ctx[g] = attn_g V_g  (Bt == Vt_g [d,m]); bf16 out
    gemm_bf16_wmma<<<dim3(blocksFor(LL, DD), GG), 256, 0, stream>>>(
        at, LL, (long long)LL * LL, Vt, LL, (long long)DD * LL,
        ctx, DD, (long long)LL * DD, LL, DD, LL, nullptr, 1);

    // 7. MLP (bias fused)
    gemm_bf16_wmma<<<dim3(blocksFor(M, 256), 1), 256, 0, stream>>>(
        ctx, DD, 0, W1T, DD, 0, h1, 256, 0, M, 256, DD, b1, 1);
    gemm_bf16_wmma<<<dim3(blocksFor(M, 128), 1), 256, 0, stream>>>(
        h1, 256, 0, W2T, 256, 0, h2, 128, 0, M, 128, 256, b2, 1);
    gemm_bf16_wmma<<<dim3(blocksFor(M, 64), 1), 256, 0, stream>>>(
        h2, 128, 0, W3T, 128, 0, h3, 64, 0, M, 64, 128, b3, 1);

    // 8. final scalar head + length special-case + mask
    score_w_kernel<<<M / 8, 256, 0, stream>>>(h3, W4, b4, len, wv);

    // 9. out[s,g] = raw[g,s,:] · w[g,:]
    weighted_sum_kernel<<<dim3(SS / 8, GG), 256, 0, stream>>>(raw, wv, out);

    (void)in_sizes; (void)n_in; (void)out_size; (void)ws_size;
}
